// Attention_55336358642806
// MI455X (gfx1250) — matrix operand order repaired and checked by reading, not compiled, not run on hardware
//
#include <hip/hip_runtime.h>

typedef __attribute__((ext_vector_type(16))) _Float16 v16h;
typedef __attribute__((ext_vector_type(2)))  _Float16 v2h;
typedef __attribute__((ext_vector_type(8)))  float    v8f;

#define WMMA_F16(a, b, c) \
  __builtin_amdgcn_wmma_f32_16x16x32_f16(false, (a), false, (b), (short)0, (c), false, false)

// ---------------- async global -> LDS copy (CDNA5 ASYNCcnt path) ----------------
#if defined(__has_builtin)
#if __has_builtin(__builtin_amdgcn_global_load_async_to_lds_b128)
#define USE_ASYNC_LDS 1
#endif
#endif
#ifndef USE_ASYNC_LDS
#define USE_ASYNC_LDS 0
#endif

typedef int v4i_vs __attribute__((vector_size(16)));

static __device__ inline void async_copy_b128(void* lds_dst, const void* gsrc) {
#if USE_ASYNC_LDS
  __builtin_amdgcn_global_load_async_to_lds_b128(
      (__attribute__((address_space(1))) v4i_vs*)(gsrc),
      (__attribute__((address_space(3))) v4i_vs*)(lds_dst), 0, 0);
#else
  *(uint4*)lds_dst = *(const uint4*)gsrc;
#endif
}

static __device__ inline void async_wait_all() {
#if USE_ASYNC_LDS
#if __has_builtin(__builtin_amdgcn_s_wait_asynccnt)
  __builtin_amdgcn_s_wait_asynccnt(0);
#else
  asm volatile("s_wait_asynccnt 0x0" ::: "memory");
#endif
#endif
}

// packed 4-byte LDS store of two adjacent f16 (dst must be 4B aligned)
static __device__ inline void store_pair(_Float16* dst, _Float16 a, _Float16 b) {
  v2h p; p[0] = a; p[1] = b;
  *(v2h*)dst = p;
}

// ---- fragment loaders (wave32, 16x16x32 f16 WMMA, same contiguous k order in A and B, not the ISA table order) ----
// All loaders below read CONTIGUOUS halves per lane -> 2x ds_load_b128 each.

// A (16xK=32) from row-major tile T[m][k]:
// lane 0-15 row M=lane, K 0-15; lane 16-31 same rows, K 16-31.
static __device__ inline v16h lds_load_a16x32(const _Float16* base, int row_base, int k_base, int pitch) {
  const int lane = threadIdx.x & 31;
  const int g = lane >> 4;
  const _Float16* p = base + (row_base + (lane & 15)) * pitch + k_base;
  v16h a;
#pragma unroll
  for (int e = 0; e < 8; ++e) a[e] = p[g * 16 + e];
#pragma unroll
  for (int e = 0; e < 8; ++e) a[8 + e] = p[g * 16 + 8 + e];
  return a;
}

// B (K=32 x 16) from an n-major tile T[n][k] (i.e. stored transposed):
// lane holds column n = n_base + (lane&15), k rows k_base + (lane>>4)*16 + e  -> contiguous.
static __device__ inline v16h lds_load_b_kmaj(const _Float16* base, int n_base, int k_base, int pitch) {
  const int lane = threadIdx.x & 31;
  const _Float16* p = base + (n_base + (lane & 15)) * pitch + k_base + (lane >> 4) * 16;
  v16h b;
#pragma unroll
  for (int e = 0; e < 16; ++e) b[e] = p[e];
  return b;
}

// ================= Kernel 1: qkv = x @ W_qkv + b_qkv  (fp32 in, f16 out) =================
// M=16384, K=256, N=768. Block tile 128x128, 256 threads = 8 waves (4 M x 2 N), 32x64 per wave.
__global__ __launch_bounds__(256) void qkv_gemm_kernel(
    const float* __restrict__ x, const float* __restrict__ Wqkv,
    const float* __restrict__ bqkv, _Float16* __restrict__ qkv) {
  __shared__ __align__(16) _Float16 As[128][40];   // [m][k], pitch 40
  __shared__ __align__(16) _Float16 Bst[128][40];  // [n][k] transposed, pitch 40
  const int t = threadIdx.x;
  const int m0 = blockIdx.x * 128;
  const int n0 = blockIdx.y * 128;
  const int lane = t & 31, w = t >> 5;
  const int mw = (w >> 1) * 32, nw = (w & 1) * 64;

  v8f z = {};
  v8f acc[2][4];
#pragma unroll
  for (int i = 0; i < 2; ++i)
#pragma unroll
    for (int j = 0; j < 4; ++j) acc[i][j] = z;

  for (int k0 = 0; k0 < 256; k0 += 32) {
    {  // stage x tile 128x32 (fp32 -> f16, packed pair stores)
      const int row = t >> 1, kc = (t & 1) * 16;
      const float* src = x + (size_t)(m0 + row) * 256 + k0 + kc;
      if (k0 + 32 < 256) __builtin_prefetch(src + 32, 0, 1);
#pragma unroll
      for (int e = 0; e < 8; ++e)
        store_pair(&As[row][kc + 2 * e], (_Float16)src[2 * e], (_Float16)src[2 * e + 1]);
    }
    {  // stage W tile 32x128 transposed: thread owns 2k x 8n sub-block
      const int kk = (t >> 4) * 2;       // 0..30
      const int nn = (t & 15) * 8;       // 0..120
      const float* r0 = Wqkv + (size_t)(k0 + kk) * 768 + n0 + nn;
      const float* r1 = r0 + 768;
      if (k0 + 32 < 256) __builtin_prefetch(r0 + (size_t)32 * 768, 0, 1);
#pragma unroll
      for (int i = 0; i < 8; ++i)
        store_pair(&Bst[nn + i][kk], (_Float16)r0[i], (_Float16)r1[i]);
    }
    __syncthreads();
    v16h a0 = lds_load_a16x32(&As[0][0], mw + 0,  0, 40);
    v16h a1 = lds_load_a16x32(&As[0][0], mw + 16, 0, 40);
#pragma unroll
    for (int j = 0; j < 4; ++j) {
      v16h bj = lds_load_b_kmaj(&Bst[0][0], nw + j * 16, 0, 40);
      acc[0][j] = WMMA_F16(a0, bj, acc[0][j]);
      acc[1][j] = WMMA_F16(a1, bj, acc[1][j]);
    }
    __syncthreads();
  }

  const int col16 = lane & 15, rb = (lane >> 4) * 8;
#pragma unroll
  for (int i = 0; i < 2; ++i)
#pragma unroll
    for (int j = 0; j < 4; ++j) {
      const int coln = n0 + nw + j * 16 + col16;
      const float bias = bqkv[coln];
#pragma unroll
      for (int jj = 0; jj < 8; ++jj) {
        const int row = m0 + mw + i * 16 + rb + jj;
        qkv[(size_t)row * 768 + coln] = (_Float16)(acc[i][j][jj] + bias);
      }
    }
}

// ================= Kernel 2: causal flash attention =================
// grid (B*H=64, S/64=16), 128 threads = 4 waves; wave w owns queries [q0+16w, q0+16w+16),
// streams K/V in 32-key tiles with online softmax; fp32 accumulators, f16 WMMA operands.
__global__ __launch_bounds__(128) void flash_attn_kernel(
    const _Float16* __restrict__ qkv, _Float16* __restrict__ att) {
  __shared__ __align__(16) _Float16 Qs[64][72];    // [q][d]
  __shared__ __align__(16) _Float16 Ks[32][72];    // [key][d]  (QK^T B-frags read rows)
  __shared__ __align__(16) _Float16 Vst[64][40];   // [d][key]  transposed (PV B-frags read rows)
  __shared__ float Ss[64][33];
  __shared__ __align__(16) _Float16 Ps[64][40];    // [q][key]
  __shared__ float Ms[64], Ls[64], Alpha[64];

  const int t = threadIdx.x;
  const int lane = t & 31, w = t >> 5;
  const int b = blockIdx.x >> 2, h = blockIdx.x & 3;
  const int q0 = blockIdx.y * 64;
  const size_t rowbase = (size_t)b * 1024;

  {  // Q tile 64x64 f16: async global->LDS copies (pure copy, no conversion)
    const int row = t >> 1, seg = t & 1;
    const _Float16* src = qkv + (rowbase + q0 + row) * 768 + h * 64 + seg * 32;
    _Float16* dst = &Qs[row][seg * 32];
#pragma unroll
    for (int e = 0; e < 4; ++e) async_copy_b128(dst + e * 8, src + e * 8);
  }
  if (t < 64) { Ms[t] = -1e30f; Ls[t] = 0.0f; }

  v8f z = {};
  v8f acc[4];
  acc[0] = z; acc[1] = z; acc[2] = z; acc[3] = z;

  const int qr = w * 16;
  const int col16 = lane & 15, rb = (lane >> 4) * 8;

  for (int k0 = 0; k0 <= q0 + 32; k0 += 32) {  // causal: last tile starts at q0+32
    __syncthreads();
    {  // stage K tile 32x64 row-major: async global->LDS
      const int row = t >> 2, seg = t & 3;
      const _Float16* ksrc = qkv + (rowbase + k0 + row) * 768 + h * 64 + 256 + seg * 16;
      if (k0 <= q0) __builtin_prefetch(ksrc + (size_t)32 * 768, 0, 1);
      _Float16* kdst = &Ks[row][seg * 16];
      async_copy_b128(kdst,     ksrc);
      async_copy_b128(kdst + 8, ksrc + 8);
    }
    {  // stage V tile 32x64 TRANSPOSED: thread owns 2key x 8d sub-block (VGPR path)
      const int kp = t >> 3;          // 0..15 -> keys 2kp, 2kp+1
      const int dg = (t & 7) * 8;     // 0..56
      const _Float16* v0p = qkv + (rowbase + k0 + 2 * kp) * 768 + h * 64 + 512 + dg;
      if (k0 <= q0) __builtin_prefetch(v0p + (size_t)32 * 768, 0, 1);
      uint4 u0 = *(const uint4*)v0p;
      uint4 u1 = *(const uint4*)(v0p + 768);
      const _Float16* h0 = (const _Float16*)&u0;
      const _Float16* h1 = (const _Float16*)&u1;
#pragma unroll
      for (int i = 0; i < 8; ++i)
        store_pair(&Vst[dg + i][2 * kp], h0[i], h1[i]);
    }
    async_wait_all();   // ASYNCcnt==0: Q (first iter) + K tile resident in LDS
    __syncthreads();

    // S = Q K^T : 16 queries x 32 keys per wave (4 WMMA)
    v8f s0 = z, s1 = z;
    {
      v16h aq0  = lds_load_a16x32(&Qs[0][0], qr, 0,  72);
      v16h aq1  = lds_load_a16x32(&Qs[0][0], qr, 32, 72);
      v16h bk00 = lds_load_b_kmaj(&Ks[0][0], 0,  0,  72);  // keys 0-15,  d 0-31
      v16h bk01 = lds_load_b_kmaj(&Ks[0][0], 0,  32, 72);  // keys 0-15,  d 32-63
      v16h bk10 = lds_load_b_kmaj(&Ks[0][0], 16, 0,  72);  // keys 16-31, d 0-31
      v16h bk11 = lds_load_b_kmaj(&Ks[0][0], 16, 32, 72);  // keys 16-31, d 32-63
      s0 = WMMA_F16(aq0, bk00, s0);
      s0 = WMMA_F16(aq1, bk01, s0);
      s1 = WMMA_F16(aq0, bk10, s1);
      s1 = WMMA_F16(aq1, bk11, s1);
    }
    // scale (1/sqrt(64)) + causal mask, park scores in LDS
#pragma unroll
    for (int jj = 0; jj < 8; ++jj) {
      const int qrow = q0 + qr + rb + jj;
      Ss[qr + rb + jj][col16]      = (k0 + col16      <= qrow) ? (float)s0[jj] * 0.125f : -1e30f;
      Ss[qr + rb + jj][16 + col16] = (k0 + 16 + col16 <= qrow) ? (float)s1[jj] * 0.125f : -1e30f;
    }
    __syncthreads();

    // online softmax: one thread per query row
    if (t < 64) {
      const float m_old = Ms[t];
      float mt = m_old;
#pragma unroll
      for (int c = 0; c < 32; ++c) mt = fmaxf(mt, Ss[t][c]);
      const float alpha = __expf(m_old - mt);
      float lsum = 0.0f;
#pragma unroll
      for (int c = 0; c < 16; ++c) {
        const float p0 = __expf(Ss[t][2 * c] - mt);
        const float p1 = __expf(Ss[t][2 * c + 1] - mt);
        store_pair(&Ps[t][2 * c], (_Float16)p0, (_Float16)p1);
        lsum += p0 + p1;
      }
      Ls[t] = Ls[t] * alpha + lsum;
      Ms[t] = mt;
      Alpha[t] = alpha;
    }
    __syncthreads();

    // rescale running accumulators, then O += P @ V (4 WMMA, K=32)
#pragma unroll
    for (int jj = 0; jj < 8; ++jj) {
      const float a = Alpha[qr + rb + jj];
      acc[0][jj] *= a; acc[1][jj] *= a; acc[2][jj] *= a; acc[3][jj] *= a;
    }
    v16h ap = lds_load_a16x32(&Ps[0][0], qr, 0, 40);
#pragma unroll
    for (int n = 0; n < 4; ++n) {
      v16h bv = lds_load_b_kmaj(&Vst[0][0], n * 16, 0, 40);  // dims n*16.., keys 0-31
      acc[n] = WMMA_F16(ap, bv, acc[n]);
    }
  }
  __syncthreads();

  // normalize by l and store f16 attention output [B*S, 256] (head-merged layout)
#pragma unroll
  for (int jj = 0; jj < 8; ++jj) {
    const int qrow = q0 + qr + rb + jj;
    const float linv = 1.0f / Ls[qr + rb + jj];
#pragma unroll
    for (int n = 0; n < 4; ++n) {
      att[(rowbase + qrow) * 256 + h * 64 + n * 16 + col16] = (_Float16)(acc[n][jj] * linv);
    }
  }
}

// ================= Kernel 3: out = att @ W_out + b_out (f16 A, fp32 out) =================
// M=16384, K=256, N=256. Block tile 128x128, same wave layout as kernel 1.
__global__ __launch_bounds__(256) void out_gemm_kernel(
    const _Float16* __restrict__ att, const float* __restrict__ Wout,
    const float* __restrict__ bout, float* __restrict__ out) {
  __shared__ __align__(16) _Float16 As[128][40];   // [m][k]
  __shared__ __align__(16) _Float16 Bst[128][40];  // [n][k] transposed
  const int t = threadIdx.x;
  const int m0 = blockIdx.x * 128;
  const int n0 = blockIdx.y * 128;
  const int lane = t & 31, w = t >> 5;
  const int mw = (w >> 1) * 32, nw = (w & 1) * 64;

  v8f z = {};
  v8f acc[2][4];
#pragma unroll
  for (int i = 0; i < 2; ++i)
#pragma unroll
    for (int j = 0; j < 4; ++j) acc[i][j] = z;

  for (int k0 = 0; k0 < 256; k0 += 32) {
    {  // stage att tile 128x32 (already f16): async global->LDS copies
      const int row = t >> 1, seg = t & 1;
      const _Float16* src = att + (size_t)(m0 + row) * 256 + k0 + seg * 16;
      if (k0 + 32 < 256) __builtin_prefetch(src + 32, 0, 1);
      _Float16* dst = &As[row][seg * 16];
      async_copy_b128(dst,     src);
      async_copy_b128(dst + 8, src + 8);
    }
    {  // stage W_out tile 32x128 transposed (fp32 -> f16 conversion, VGPR path)
      const int kk = (t >> 4) * 2;
      const int nn = (t & 15) * 8;
      const float* r0 = Wout + (size_t)(k0 + kk) * 256 + n0 + nn;
      const float* r1 = r0 + 256;
      if (k0 + 32 < 256) __builtin_prefetch(r0 + (size_t)32 * 256, 0, 1);
#pragma unroll
      for (int i = 0; i < 8; ++i)
        store_pair(&Bst[nn + i][kk], (_Float16)r0[i], (_Float16)r1[i]);
    }
    async_wait_all();   // ASYNCcnt==0: A tile resident in LDS
    __syncthreads();
    v16h a0 = lds_load_a16x32(&As[0][0], mw + 0,  0, 40);
    v16h a1 = lds_load_a16x32(&As[0][0], mw + 16, 0, 40);
#pragma unroll
    for (int j = 0; j < 4; ++j) {
      v16h bj = lds_load_b_kmaj(&Bst[0][0], nw + j * 16, 0, 40);
      acc[0][j] = WMMA_F16(a0, bj, acc[0][j]);
      acc[1][j] = WMMA_F16(a1, bj, acc[1][j]);
    }
    __syncthreads();
  }

  const int col16 = lane & 15, rb = (lane >> 4) * 8;
#pragma unroll
  for (int i = 0; i < 2; ++i)
#pragma unroll
    for (int j = 0; j < 4; ++j) {
      const int coln = n0 + nw + j * 16 + col16;
      const float bias = bout[coln];
#pragma unroll
      for (int jj = 0; jj < 8; ++jj) {
        const int row = m0 + mw + i * 16 + rb + jj;
        out[(size_t)row * 256 + coln] = acc[i][j][jj] + bias;
      }
    }
}

extern "C" void kernel_launch(void* const* d_in, const int* in_sizes, int n_in,
                              void* d_out, int out_size, void* d_ws, size_t ws_size,
                              hipStream_t stream) {
  (void)in_sizes; (void)n_in; (void)out_size; (void)ws_size;
  const float* x    = (const float*)d_in[0];
  const float* Wqkv = (const float*)d_in[1];
  const float* bqkv = (const float*)d_in[2];
  const float* Wout = (const float*)d_in[3];
  const float* bout = (const float*)d_in[4];
  float* out = (float*)d_out;

  // workspace: qkv f16 [16384 x 768] (25.2 MB) then att f16 [16384 x 256] (8.4 MB) = 32 MB
  _Float16* qkv = (_Float16*)d_ws;
  _Float16* att = qkv + (size_t)16384 * 768;

  qkv_gemm_kernel<<<dim3(128, 6), 256, 0, stream>>>(x, Wqkv, bqkv, qkv);
  flash_attn_kernel<<<dim3(64, 16), 128, 0, stream>>>(qkv, att);
  out_gemm_kernel<<<dim3(128, 2), 256, 0, stream>>>(att, Wout, bout, out);
}
